// GCN_28467043238508
// MI455X (gfx1250) — compile-verified, matrix operand order checked by reading, not run on hardware
//
#include <hip/hip_runtime.h>
#include <hip/hip_bf16.h>
#include <math.h>

// Problem constants (fixed by the reference)
#define N_NODES 50000
#define N_EDGES 800000
#define F_IN    64
#define D_HID   128
#define B_GRAPH 512
#define O_OUT   138
#define SCALE_F 0.08838834764831845f  // 1/sqrt(128)

typedef __attribute__((ext_vector_type(16))) __bf16 v16bf;
typedef __attribute__((ext_vector_type(8)))  float  v8f;

// ---------- monotone float<->uint encoding for atomic max on floats ----------
__device__ __forceinline__ unsigned f2mono(float f) {
    unsigned u = __float_as_uint(f);
    return (u & 0x80000000u) ? ~u : (u | 0x80000000u);
}
__device__ __forceinline__ float mono2f(unsigned u) {
    u = (u & 0x80000000u) ? (u & 0x7fffffffu) : ~u;
    return __uint_as_float(u);
}

// ---------- utility kernels ----------
__global__ void fill_u32(unsigned* __restrict__ p, unsigned v, int total) {
    int i = blockIdx.x * blockDim.x + threadIdx.x;
    if (i < total) p[i] = v;
}

// f32 [stk, kdim, 128] -> bf16 [stk, 128, kdim] (transpose last two dims)
__global__ void wconv(const float* __restrict__ W, __bf16* __restrict__ Wt,
                      int kdim, int total) {
    int i = blockIdx.x * blockDim.x + threadIdx.x;
    if (i >= total) return;
    int per = kdim * D_HID;
    int l = i / per, r = i % per;
    int k = r / D_HID, n = r % D_HID;
    Wt[(size_t)l * per + (size_t)n * kdim + k] = (__bf16)W[i];
}

// optional SELU, then f32 -> bf16
__global__ void conv_selu(const float* __restrict__ h, __bf16* __restrict__ hbf,
                          int total, int apply) {
    int i = blockIdx.x * blockDim.x + threadIdx.x;
    if (i >= total) return;
    float x = h[i];
    if (apply) {
        const float kS = 1.0507009873554805f, kA = 1.6732632423543772f;
        x = (x > 0.f) ? kS * x : kS * kA * (__expf(x) - 1.f);
    }
    hbf[i] = (__bf16)x;
}

// ---------- fused 4-projection WMMA GEMM: out = hbf @ W + b ----------
// grid.x = N/16 (row tiles), grid.y = 4 (which projection), 256 thr = 8 waves,
// each wave owns one 16-column tile of the 128 output columns.
__global__ __launch_bounds__(256)
void gemm4_wmma(const __bf16* __restrict__ hbf, int kdim,
                const __bf16* __restrict__ Wt0, const float* __restrict__ b0, float* __restrict__ o0,
                const __bf16* __restrict__ Wt1, const float* __restrict__ b1, float* __restrict__ o1,
                const __bf16* __restrict__ Wt2, const float* __restrict__ b2, float* __restrict__ o2,
                const __bf16* __restrict__ Wt3, const float* __restrict__ b3, float* __restrict__ o3) {
    __shared__ __align__(16) __bf16 lds_a[16 * D_HID];

    const int tid  = threadIdx.x;
    const int row0 = blockIdx.x * 16;

    // cooperative A-tile load: 16 rows x kdim bf16, contiguous in global
    const int nvec = (16 * kdim) / 8;  // uint4 carries 8 bf16
    if (tid < nvec)
        reinterpret_cast<uint4*>(lds_a)[tid] =
            reinterpret_cast<const uint4*>(hbf + (size_t)row0 * kdim)[tid];
    __syncthreads();

    const __bf16* Wt; const float* bias; float* out;
    switch (blockIdx.y) {
        case 0:  Wt = Wt0; bias = b0; out = o0; break;
        case 1:  Wt = Wt1; bias = b1; out = o1; break;
        case 2:  Wt = Wt2; bias = b2; out = o2; break;
        default: Wt = Wt3; bias = b3; out = o3; break;
    }

    const int wave  = tid >> 5;
    const int lane  = tid & 31;
    const int nbase = wave * 16;
    const int n     = nbase + (lane & 15);
    const int m     = lane & 15;
    const int hi    = (lane >> 4) & 1;  // lane-half selects K sub-blocks

    union AB { v16bf v; uint4 u[2]; };
    v8f acc = {};
    for (int k0 = 0; k0 < kdim; k0 += 32) {
        const int kLo = k0 + hi * 8;       // frag elems 0..7  -> K = kLo..kLo+7
        const int kHi = k0 + 16 + hi * 8;  // frag elems 8..15 -> K = kHi..kHi+7
        AB a, b;
        a.u[0] = *reinterpret_cast<const uint4*>(&lds_a[m * kdim + kLo]);
        a.u[1] = *reinterpret_cast<const uint4*>(&lds_a[m * kdim + kHi]);
        b.u[0] = *reinterpret_cast<const uint4*>(&Wt[(size_t)n * kdim + kLo]);
        b.u[1] = *reinterpret_cast<const uint4*>(&Wt[(size_t)n * kdim + kHi]);
        acc = __builtin_amdgcn_wmma_f32_16x16x32_bf16(false, a.v, false, b.v,
                                                      (short)0, acc, false, false);
    }
    const float bn = bias[n];
#pragma unroll
    for (int r = 0; r < 8; ++r) {
        const int mr = row0 + (hi ? r + 8 : r);
        out[(size_t)mr * D_HID + n] = acc[r] + bn;
    }
}

// ---------- edge kernels: one wave32 per edge, 4 floats per lane ----------
__global__ __launch_bounds__(256)
void edge_score(const float* __restrict__ q, const float* __restrict__ kk,
                const int* __restrict__ src, const int* __restrict__ dst,
                float* __restrict__ score, unsigned* __restrict__ mbits) {
    int e = blockIdx.x * 8 + (threadIdx.x >> 5);
    if (e >= N_EDGES) return;
    int lane = threadIdx.x & 31;
    int s = src[e], d = dst[e];
    float4 qv = reinterpret_cast<const float4*>(q  + (size_t)d * D_HID)[lane];
    float4 kv = reinterpret_cast<const float4*>(kk + (size_t)s * D_HID)[lane];
    float p = qv.x * kv.x + qv.y * kv.y + qv.z * kv.z + qv.w * kv.w;
    p += __shfl_xor(p, 16);
    p += __shfl_xor(p, 8);
    p += __shfl_xor(p, 4);
    p += __shfl_xor(p, 2);
    p += __shfl_xor(p, 1);
    if (lane == 0) {
        float sc = p * SCALE_F;
        score[e] = sc;
        atomicMax(&mbits[d], f2mono(sc));
    }
}

__global__ void edge_exp(float* __restrict__ score, const int* __restrict__ dst,
                         const unsigned* __restrict__ mbits, float* __restrict__ z) {
    int e = blockIdx.x * blockDim.x + threadIdx.x;
    if (e >= N_EDGES) return;
    int d = dst[e];
    float ee = __expf(score[e] - mono2f(mbits[d]));
    score[e] = ee;
    atomicAdd(&z[d], ee);
}

__global__ __launch_bounds__(256)
void edge_agg(const float* __restrict__ v, const float* __restrict__ escore,
              const float* __restrict__ z, const int* __restrict__ src,
              const int* __restrict__ dst, float* __restrict__ hnext) {
    int e = blockIdx.x * 8 + (threadIdx.x >> 5);
    if (e >= N_EDGES) return;
    int lane = threadIdx.x & 31;
    int s = src[e], d = dst[e];
    float alpha = escore[e] / z[d];
    float4 vv = reinterpret_cast<const float4*>(v + (size_t)s * D_HID)[lane];
    float* o = hnext + (size_t)d * D_HID + lane * 4;
    atomicAdd(o + 0, alpha * vv.x);
    atomicAdd(o + 1, alpha * vv.y);
    atomicAdd(o + 2, alpha * vv.z);
    atomicAdd(o + 3, alpha * vv.w);
}

// ---------- pooling + head ----------
__global__ void pool_max(const float* __restrict__ h, const int* __restrict__ batch,
                         unsigned* __restrict__ pbits, int total) {
    int i = blockIdx.x * blockDim.x + threadIdx.x;
    if (i >= total) return;
    int n = i >> 7, dcol = i & 127;
    atomicMax(&pbits[(size_t)batch[n] * D_HID + dcol], f2mono(h[i]));
}

__global__ __launch_bounds__(160)
void final_lin(const unsigned* __restrict__ pbits, const float* __restrict__ Wlin,
               const float* __restrict__ blin, float* __restrict__ out) {
    __shared__ float prow[D_HID];
    int b = blockIdx.x, t = threadIdx.x;
    if (t < D_HID) prow[t] = mono2f(pbits[(size_t)b * D_HID + t]);
    __syncthreads();
    if (t < O_OUT) {
        float acc = blin[t];
        for (int d = 0; d < D_HID; ++d) acc += prow[d] * Wlin[d * O_OUT + t];
        out[(size_t)b * O_OUT + t] = 1.f / (1.f + __expf(-acc));
    }
}

// ---------- host orchestration ----------
extern "C" void kernel_launch(void* const* d_in, const int* in_sizes, int n_in,
                              void* d_out, int out_size, void* d_ws, size_t ws_size,
                              hipStream_t stream) {
    const float* x     = (const float*)d_in[0];
    const int*   src   = (const int*)d_in[1];
    const int*   dst   = (const int*)d_in[2];
    const int*   batch = (const int*)d_in[3];
    const float* Wq1 = (const float*)d_in[4],  *bq1 = (const float*)d_in[5];
    const float* Wk1 = (const float*)d_in[6],  *bk1 = (const float*)d_in[7];
    const float* Wv1 = (const float*)d_in[8],  *bv1 = (const float*)d_in[9];
    const float* Ws1 = (const float*)d_in[10], *bs1 = (const float*)d_in[11];
    const float* Wq  = (const float*)d_in[12], *bq  = (const float*)d_in[13];
    const float* Wk  = (const float*)d_in[14], *bk  = (const float*)d_in[15];
    const float* Wv  = (const float*)d_in[16], *bv  = (const float*)d_in[17];
    const float* Ws  = (const float*)d_in[18], *bs  = (const float*)d_in[19];
    const float* Wlin = (const float*)d_in[20], *blin = (const float*)d_in[21];
    float* out = (float*)d_out;

    // workspace carve-up (~150 MB total)
    char* p = (char*)d_ws;
    auto alloc = [&](size_t bytes) -> char* {
        char* r = p; p += (bytes + 255) & ~(size_t)255; return r;
    };
    __bf16* hbf   = (__bf16*)alloc((size_t)N_NODES * D_HID * 2);
    float*  hA    = (float*)alloc((size_t)N_NODES * D_HID * 4);
    float*  hB    = (float*)alloc((size_t)N_NODES * D_HID * 4);
    float*  qbuf  = (float*)alloc((size_t)N_NODES * D_HID * 4);
    float*  kbuf  = (float*)alloc((size_t)N_NODES * D_HID * 4);
    float*  vbuf  = (float*)alloc((size_t)N_NODES * D_HID * 4);
    float*  score = (float*)alloc((size_t)N_EDGES * 4);
    unsigned* mbits = (unsigned*)alloc((size_t)N_NODES * 4);
    float*  z     = (float*)alloc((size_t)N_NODES * 4);
    unsigned* pbits = (unsigned*)alloc((size_t)B_GRAPH * D_HID * 4);
    __bf16* Wt1q = (__bf16*)alloc((size_t)F_IN * D_HID * 2);
    __bf16* Wt1k = (__bf16*)alloc((size_t)F_IN * D_HID * 2);
    __bf16* Wt1v = (__bf16*)alloc((size_t)F_IN * D_HID * 2);
    __bf16* Wt1s = (__bf16*)alloc((size_t)F_IN * D_HID * 2);
    __bf16* Wtsq = (__bf16*)alloc((size_t)7 * D_HID * D_HID * 2);
    __bf16* Wtsk = (__bf16*)alloc((size_t)7 * D_HID * D_HID * 2);
    __bf16* Wtsv = (__bf16*)alloc((size_t)7 * D_HID * D_HID * 2);
    __bf16* Wtss = (__bf16*)alloc((size_t)7 * D_HID * D_HID * 2);

    // weight convert + transpose (bf16, [n][k] layout)
    {
        int t1 = F_IN * D_HID, ts = 7 * D_HID * D_HID;
        wconv<<<(t1 + 255) / 256, 256, 0, stream>>>(Wq1, Wt1q, F_IN, t1);
        wconv<<<(t1 + 255) / 256, 256, 0, stream>>>(Wk1, Wt1k, F_IN, t1);
        wconv<<<(t1 + 255) / 256, 256, 0, stream>>>(Wv1, Wt1v, F_IN, t1);
        wconv<<<(t1 + 255) / 256, 256, 0, stream>>>(Ws1, Wt1s, F_IN, t1);
        wconv<<<(ts + 255) / 256, 256, 0, stream>>>(Wq, Wtsq, D_HID, ts);
        wconv<<<(ts + 255) / 256, 256, 0, stream>>>(Wk, Wtsk, D_HID, ts);
        wconv<<<(ts + 255) / 256, 256, 0, stream>>>(Wv, Wtsv, D_HID, ts);
        wconv<<<(ts + 255) / 256, 256, 0, stream>>>(Ws, Wtss, D_HID, ts);
    }

    float* h = hA;   // current layer input (f32), valid from layer 1 on
    float* hn = hB;  // next layer output
    const int edgeBlocks = (N_EDGES + 7) / 8;

    for (int L = 0; L < 8; ++L) {
        const int kd = (L == 0) ? F_IN : D_HID;
        // 1) input -> bf16 (SELU for layers 2..8)
        if (L == 0) {
            int tot = N_NODES * F_IN;
            conv_selu<<<(tot + 255) / 256, 256, 0, stream>>>(x, hbf, tot, 0);
        } else {
            int tot = N_NODES * D_HID;
            conv_selu<<<(tot + 255) / 256, 256, 0, stream>>>(h, hbf, tot, 1);
        }
        // 2) projections: q,k,v and skip (skip written directly into hn)
        const __bf16 *wq, *wk, *wv, *ws;
        const float *pbq, *pbk, *pbv, *pbs;
        if (L == 0) {
            wq = Wt1q; wk = Wt1k; wv = Wt1v; ws = Wt1s;
            pbq = bq1; pbk = bk1; pbv = bv1; pbs = bs1;
        } else {
            size_t off = (size_t)(L - 1) * D_HID * D_HID;
            wq = Wtsq + off; wk = Wtsk + off; wv = Wtsv + off; ws = Wtss + off;
            pbq = bq + (L - 1) * D_HID; pbk = bk + (L - 1) * D_HID;
            pbv = bv + (L - 1) * D_HID; pbs = bs + (L - 1) * D_HID;
        }
        dim3 g(N_NODES / 16, 4);
        gemm4_wmma<<<g, 256, 0, stream>>>(hbf, kd,
                                          wq, pbq, qbuf,
                                          wk, pbk, kbuf,
                                          wv, pbv, vbuf,
                                          ws, pbs, hn);
        // 3) segment softmax state init
        fill_u32<<<(N_NODES + 255) / 256, 256, 0, stream>>>(mbits, 0u, N_NODES);
        fill_u32<<<(N_NODES + 255) / 256, 256, 0, stream>>>((unsigned*)z, 0u, N_NODES);
        // 4) attention scores + segment max
        edge_score<<<edgeBlocks, 256, 0, stream>>>(qbuf, kbuf, src, dst, score, mbits);
        // 5) exp + segment sum
        edge_exp<<<(N_EDGES + 255) / 256, 256, 0, stream>>>(score, dst, mbits, z);
        // 6) weighted aggregation into hn (on top of the skip projection)
        edge_agg<<<edgeBlocks, 256, 0, stream>>>(vbuf, score, z, src, dst, hn);
        // swap
        float* t = h; h = hn; hn = t;
    }

    // global max pool + head
    fill_u32<<<(B_GRAPH * D_HID + 255) / 256, 256, 0, stream>>>(pbits, 0u, B_GRAPH * D_HID);
    pool_max<<<(N_NODES * D_HID + 255) / 256, 256, 0, stream>>>(h, batch, pbits, N_NODES * D_HID);
    final_lin<<<B_GRAPH, 160, 0, stream>>>(pbits, Wlin, blin, out);

    (void)in_sizes; (void)n_in; (void)out_size; (void)ws_size;
}
